// RGATModel_v3_2_20151986553246
// MI455X (gfx1250) — matrix operand order repaired and checked by reading, not compiled, not run on hardware
//
#include <hip/hip_runtime.h>

typedef __attribute__((ext_vector_type(16))) _Float16 v16h;
typedef __attribute__((ext_vector_type(8)))  float    v8f;

#define NEG_SLOPE 0.2f
#define BN_EPS    1e-5f
#define AGG_LD    16   // stride for agg / pre-BN h buffers
#define HELU_LD   32   // stride for post-ELU activations (zero-padded K for WMMA)

// ---------------------------------------------------------------------------
// Pack weight matrix into the exact WMMA B-operand register layout:
//   bp[((tn*KSTEPS + ks)*32 + lane)*16 + i] = Wcat[k0 + (i&7) + (lane>=16)*8 + (i>=8)*16, tn*16 + lane%16]
// Wcat[k, rel*O + o] = W[rel*KW*O + k*O + o]; out-of-range (rel>=R or k>=KW) = 0.
// All divergent index math happens once here, off the GEMM hot path.
// ---------------------------------------------------------------------------
template<int K, int O, int R, int TILES_N>
__global__ void k_pack_w(const float* __restrict__ W, _Float16* __restrict__ bp, int KW)
{
    constexpr int KSTEPS = K / 32;
    constexpr int TOTAL  = TILES_N * KSTEPS * 32 * 16;
    const int idx = blockIdx.x * blockDim.x + threadIdx.x;
    if (idx >= TOTAL) return;
    const int i    = idx & 15;
    const int lane = (idx >> 4) & 31;
    const int ks   = (idx >> 9) % KSTEPS;
    const int tn   = idx / (KSTEPS * 512);
    const int g    = lane >> 4;
    const int r16  = lane & 15;
    const int c    = tn * 16 + r16;
    const int rel  = c / O;
    const int o    = c - rel * O;
    const int kb   = ks * 32 + (i & 7) + (g << 3) + ((i >> 3) << 4);
    const float v  = (rel < R && kb < KW) ? W[((size_t)rel * KW + kb) * O + o] : 0.0f;
    bp[idx] = (_Float16)v;
}

// ---------------------------------------------------------------------------
// WMMA f16 GEMM: XT[n, c] = sum_k X[n,k] * Wcat[k,c].  One wave = 16x16 tile.
// Requires N % 16 == 0 (50000 = 3125*16) and K % 32 == 0 (X rows padded).
// A fill: 4 contiguous float4 loads + packed f32->f16 converts (no branches).
// B fill: one v16h load per lane from the pre-packed panel (lives in L2).
// ---------------------------------------------------------------------------
template<int K>
__global__ __launch_bounds__(256)
void k_wmma_xform(const float* __restrict__ X, int ldx,
                  const _Float16* __restrict__ Bpack,
                  float* __restrict__ XT, int ldxt,
                  int tiles_m, int tiles_n)
{
    static_assert(K % 32 == 0, "K must be padded to a multiple of 32");
    constexpr int KSTEPS = K / 32;

    const int wave = blockIdx.x * (blockDim.x >> 5) + (threadIdx.x >> 5);
    if (wave >= tiles_m * tiles_n) return;   // wave-uniform: EXEC stays all-ones
    const int lane = threadIdx.x & 31;
    const int g    = lane >> 4;
    const int r16  = lane & 15;
    const int tn   = wave % tiles_n;
    const int tm   = wave / tiles_n;

    const float* __restrict__ xrow = X + (size_t)(tm * 16 + r16) * ldx;
    const v16h*  __restrict__ bvp  = (const v16h*)Bpack;

    v8f acc = {};
#pragma unroll
    for (int ks = 0; ks < KSTEPS; ++ks) {
        // A (16x32 f16): lanes 0-15 hold K {k0..k0+7, k0+16..k0+23};
        //                lanes 16-31 hold K {k0+8..k0+15, k0+24..k0+31}.
        const float4* __restrict__ x0 = (const float4*)(xrow + ks * 32 + (g << 3));
        const float4* __restrict__ x1 = (const float4*)(xrow + ks * 32 + 16 + (g << 3));
        const float4 p0 = x0[0], p1 = x0[1];
        const float4 p2 = x1[0], p3 = x1[1];
        v16h a;
        a[0]  = (_Float16)p0.x; a[1]  = (_Float16)p0.y;
        a[2]  = (_Float16)p0.z; a[3]  = (_Float16)p0.w;
        a[4]  = (_Float16)p1.x; a[5]  = (_Float16)p1.y;
        a[6]  = (_Float16)p1.z; a[7]  = (_Float16)p1.w;
        a[8]  = (_Float16)p2.x; a[9]  = (_Float16)p2.y;
        a[10] = (_Float16)p2.z; a[11] = (_Float16)p2.w;
        a[12] = (_Float16)p3.x; a[13] = (_Float16)p3.y;
        a[14] = (_Float16)p3.z; a[15] = (_Float16)p3.w;

        const v16h b = bvp[(tn * KSTEPS + ks) * 32 + lane];

        acc = __builtin_amdgcn_wmma_f32_16x16x32_f16(false, a, false, b,
                                                     (short)0, acc, false, false);
    }
    // D (16x16 f32): VGPR j -> row = j + 8*g, column r16 within tile.
    const int c = tn * 16 + r16;
#pragma unroll
    for (int j = 0; j < 8; ++j)
        XT[(size_t)(tm * 16 + (g << 3) + j) * ldxt + c] = acc[j];
}

// ---------------------------------------------------------------------------
// Per-call accumulator init (graph-replay safe)
// ---------------------------------------------------------------------------
__global__ void k_init(float* __restrict__ agg, float* __restrict__ denom,
                       float* __restrict__ amax, float* __restrict__ stats, int N)
{
    const int i = blockIdx.x * blockDim.x + threadIdx.x;
    if (i < N * AGG_LD) agg[i] = 0.0f;
    if (i < N) { denom[i] = 0.0f; amax[i] = -__builtin_inff(); }
    if (i < 2 * AGG_LD) stats[i] = 0.0f;
}

__device__ inline void atomic_max_f32(float* addr, float val)
{
    unsigned int* ia  = (unsigned int*)addr;
    unsigned int  old = *ia;
    while (__uint_as_float(old) < val) {
        const unsigned int assumed = old;
        old = atomicCAS(ia, assumed, __float_as_uint(val));
        if (old == assumed) break;
    }
}

// ---------------------------------------------------------------------------
// Edge pass 1: logits + leaky_relu + segment max over dst
// ---------------------------------------------------------------------------
template<int O>
__global__ void k_alpha(const float* __restrict__ XT, int ldxt,
                        const int* __restrict__ srcI, const int* __restrict__ dstI,
                        const int* __restrict__ etype, const float* __restrict__ eattr,
                        const float* __restrict__ q,  const float* __restrict__ kv,
                        const float* __restrict__ We, const float* __restrict__ ev,
                        float* __restrict__ alpha, float* __restrict__ amax, int E)
{
    const int e = blockIdx.x * blockDim.x + threadIdx.x;
    if (e >= E) return;
    const int s = srcI[e], d = dstI[e], t = etype[e];
    const float* __restrict__ xi = XT + (size_t)d * ldxt + t * O;
    const float* __restrict__ xj = XT + (size_t)s * ldxt + t * O;
    float ai = 0.0f, aj = 0.0f, we = 0.0f;
#pragma unroll
    for (int o = 0; o < O; ++o) {
        ai += xi[o] * q[o];
        aj += xj[o] * kv[o];
        we += We[o] * ev[o];
    }
    float a = ai + aj + eattr[e] * we;
    a = (a > 0.0f) ? a : NEG_SLOPE * a;
    alpha[e] = a;
    atomic_max_f32(&amax[d], a);
}

// ---------------------------------------------------------------------------
// Edge pass 2: exp(alpha - max) + segment sum over dst
// ---------------------------------------------------------------------------
__global__ void k_exp(const float* __restrict__ alpha, const float* __restrict__ amax,
                      const int* __restrict__ dstI, float* __restrict__ exb,
                      float* __restrict__ denom, int E)
{
    const int e = blockIdx.x * blockDim.x + threadIdx.x;
    if (e >= E) return;
    const int d = dstI[e];
    const float ex = __expf(alpha[e] - amax[d]);
    exb[e] = ex;
    atomicAdd(&denom[d], ex);
}

// ---------------------------------------------------------------------------
// Edge pass 3: normalize + weighted scatter-add of source features
// ---------------------------------------------------------------------------
template<int O>
__global__ void k_scatter(const float* __restrict__ exb, const float* __restrict__ denom,
                          const float* __restrict__ XT, int ldxt,
                          const int* __restrict__ srcI, const int* __restrict__ dstI,
                          const int* __restrict__ etype,
                          float* __restrict__ agg, int E)
{
    const int e = blockIdx.x * blockDim.x + threadIdx.x;
    if (e >= E) return;
    const int s = srcI[e], d = dstI[e], t = etype[e];
    const float w = exb[e] / (denom[d] + 1e-16f);
    const float* __restrict__ xj = XT + (size_t)s * ldxt + t * O;
    float* __restrict__ ad = agg + (size_t)d * AGG_LD;
#pragma unroll
    for (int o = 0; o < O; ++o) atomicAdd(&ad[o], w * xj[o]);
}

// ---------------------------------------------------------------------------
// h = agg + bias; per-column sum/sumsq via LDS ds_add_f32 reduce -> global
// ---------------------------------------------------------------------------
__global__ void k_bias_stats(const float* __restrict__ agg, const float* __restrict__ b,
                             int O, float* __restrict__ h, float* __restrict__ stats, int N)
{
    __shared__ float ssum[AGG_LD], ssq[AGG_LD];
    if (threadIdx.x < AGG_LD) { ssum[threadIdx.x] = 0.0f; ssq[threadIdx.x] = 0.0f; }
    __syncthreads();
    const int n = blockIdx.x * blockDim.x + threadIdx.x;
    if (n < N) {
        for (int o = 0; o < O; ++o) {
            const float v = agg[(size_t)n * AGG_LD + o] + b[o];
            h[(size_t)n * AGG_LD + o] = v;
            atomicAdd(&ssum[o], v);
            atomicAdd(&ssq[o], v * v);
        }
    }
    __syncthreads();
    if (threadIdx.x < AGG_LD) {
        atomicAdd(&stats[threadIdx.x], ssum[threadIdx.x]);
        atomicAdd(&stats[AGG_LD + threadIdx.x], ssq[threadIdx.x]);
    }
}

// ---------------------------------------------------------------------------
// Training-mode BN (biased variance) + ELU; writes stride-ld_out rows with
// columns O..ld_out-1 zeroed so the next WMMA A-load needs no bounds checks.
// ---------------------------------------------------------------------------
__global__ void k_bn_elu(const float* __restrict__ h, const float* __restrict__ stats,
                         const float* __restrict__ g, const float* __restrict__ beta,
                         int O, int N, float* __restrict__ out, int ld_out)
{
    const int n = blockIdx.x * blockDim.x + threadIdx.x;
    if (n >= N) return;
    const float invN = 1.0f / (float)N;
    for (int o = 0; o < O; ++o) {
        const float mu  = stats[o] * invN;
        const float var = stats[AGG_LD + o] * invN - mu * mu;
        float v = (h[(size_t)n * AGG_LD + o] - mu) * rsqrtf(var + BN_EPS);
        v = v * g[o] + beta[o];
        out[(size_t)n * ld_out + o] = (v > 0.0f) ? v : (__expf(v) - 1.0f);
    }
    for (int o = O; o < ld_out; ++o) out[(size_t)n * ld_out + o] = 0.0f;
}

// ---------------------------------------------------------------------------
// Head: out[n] = h[n,:] . w_head + b_head
// ---------------------------------------------------------------------------
__global__ void k_head(const float* __restrict__ h, int ldh,
                       const float* __restrict__ wh, const float* __restrict__ bh,
                       float* __restrict__ out, int O, int N)
{
    const int n = blockIdx.x * blockDim.x + threadIdx.x;
    if (n >= N) return;
    float acc = bh[0];
    for (int o = 0; o < O; ++o) acc += h[(size_t)n * ldh + o] * wh[o];
    out[n] = acc;
}

// ---------------------------------------------------------------------------
extern "C" void kernel_launch(void* const* d_in, const int* in_sizes, int n_in,
                              void* d_out, int out_size, void* d_ws, size_t ws_size,
                              hipStream_t stream)
{
    const float* x   = (const float*)d_in[0];
    const int*   ei  = (const int*)  d_in[1];
    const int*   et  = (const int*)  d_in[2];
    const float* ea  = (const float*)d_in[3];
    const float* W1  = (const float*)d_in[4];
    const float* q1  = (const float*)d_in[5];
    const float* k1  = (const float*)d_in[6];
    const float* e1  = (const float*)d_in[7];
    const float* We1 = (const float*)d_in[8];
    const float* b1  = (const float*)d_in[9];
    const float* W2  = (const float*)d_in[10];
    const float* q2  = (const float*)d_in[11];
    const float* k2  = (const float*)d_in[12];
    const float* e2  = (const float*)d_in[13];
    const float* We2 = (const float*)d_in[14];
    const float* b2  = (const float*)d_in[15];
    const float* g1  = (const float*)d_in[16];
    const float* be1 = (const float*)d_in[17];
    const float* g2  = (const float*)d_in[18];
    const float* be2 = (const float*)d_in[19];
    const float* wh  = (const float*)d_in[20];
    const float* bh  = (const float*)d_in[21];
    float* out = (float*)d_out;

    const int F = 128;
    const int N = in_sizes[0] / F;     // 50000
    const int E = in_sizes[2];         // 1.6M
    const int* srcI = ei;
    const int* dstI = ei + E;

    // ---- workspace carve (floats, 64-float = 256B aligned blocks) ----------
    float* ws  = (float*)d_ws;
    size_t off = 0;
    auto carve = [&](size_t nflt) { float* p = ws + off; off += (nflt + 63) & ~(size_t)63; return p; };
    float*     xt    = carve((size_t)N * 64);       // relation transform [N,64]
    float*     alpha = carve((size_t)E);            // logits
    float*     exb   = carve((size_t)E);            // exp(logit - max)
    float*     amax  = carve((size_t)N);            // segment max
    float*     denom = carve((size_t)N);            // segment sum
    float*     agg   = carve((size_t)N * AGG_LD);   // scatter target
    float*     hbuf  = carve((size_t)N * AGG_LD);   // pre-BN h
    float*     helu  = carve((size_t)N * HELU_LD);  // post-ELU h (K-padded)
    float*     stats = carve(2 * AGG_LD);           // column sum/sumsq
    _Float16*  bp1   = (_Float16*)carve(4 * 4 * 512 / 2);  // packed B, layer 1
    _Float16*  bp2   = (_Float16*)carve(3 * 1 * 512 / 2);  // packed B, layer 2
    (void)ws_size; (void)n_in; (void)out_size;

    const dim3 blk(256);
    const int tiles_m    = (N + 15) / 16;                // 3125 (N % 16 == 0)
    const int initBlocks = (N * AGG_LD + 255) / 256;
    const int edgeBlocks = (E + 255) / 256;
    const int nodeBlocks = (N + 255) / 256;

    // ======================= Layer 1 (K=128 -> R*15, pad 64) ================
    {
        const int tiles_n = 4;                            // ceil(60/16)
        k_pack_w<128, 15, 4, 4><<<(4 * 4 * 512 + 255) / 256, blk, 0, stream>>>(W1, bp1, 128);
        const int blocks = (tiles_m * tiles_n + 7) / 8;   // 8 waves / block
        k_wmma_xform<128><<<blocks, blk, 0, stream>>>(x, F, bp1, xt, 64, tiles_m, tiles_n);
    }
    k_init<<<initBlocks, blk, 0, stream>>>(agg, denom, amax, stats, N);
    k_alpha<15><<<edgeBlocks, blk, 0, stream>>>(xt, 64, srcI, dstI, et, ea,
                                                q1, k1, We1, e1, alpha, amax, E);
    k_exp<<<edgeBlocks, blk, 0, stream>>>(alpha, amax, dstI, exb, denom, E);
    k_scatter<15><<<edgeBlocks, blk, 0, stream>>>(exb, denom, xt, 64,
                                                  srcI, dstI, et, agg, E);
    k_bias_stats<<<nodeBlocks, blk, 0, stream>>>(agg, b1, 15, hbuf, stats, N);
    k_bn_elu<<<nodeBlocks, blk, 0, stream>>>(hbuf, stats, g1, be1, 15, N, helu, HELU_LD);

    // ======================= Layer 2 (K=15 pad 32 -> R*10, pad 48) ==========
    {
        const int tiles_n = 3;                            // ceil(40/16)
        k_pack_w<32, 10, 4, 3><<<(3 * 1 * 512 + 255) / 256, blk, 0, stream>>>(W2, bp2, 15);
        const int blocks = (tiles_m * tiles_n + 7) / 8;
        k_wmma_xform<32><<<blocks, blk, 0, stream>>>(helu, HELU_LD, bp2, xt, 64, tiles_m, tiles_n);
    }
    k_init<<<initBlocks, blk, 0, stream>>>(agg, denom, amax, stats, N);
    k_alpha<10><<<edgeBlocks, blk, 0, stream>>>(xt, 64, srcI, dstI, et, ea,
                                                q2, k2, We2, e2, alpha, amax, E);
    k_exp<<<edgeBlocks, blk, 0, stream>>>(alpha, amax, dstI, exb, denom, E);
    k_scatter<10><<<edgeBlocks, blk, 0, stream>>>(exb, denom, xt, 64,
                                                  srcI, dstI, et, agg, E);
    k_bias_stats<<<nodeBlocks, blk, 0, stream>>>(agg, b2, 10, hbuf, stats, N);
    k_bn_elu<<<nodeBlocks, blk, 0, stream>>>(hbuf, stats, g2, be2, 10, N, helu, HELU_LD);

    // ======================= Head ==========================================
    k_head<<<nodeBlocks, blk, 0, stream>>>(helu, HELU_LD, wh, bh, out, 10, N);
}
